// AttendAndSpell_21509196218681
// MI455X (gfx1250) — matrix operand order repaired and checked by reading, not compiled, not run on hardware
//
#include <hip/hip_runtime.h>
#include <hip/hip_bf16.h>
#include <math.h>

#define HID 512
#define NB  128   // batch
#define NR  512   // attention memory rows
#define NT  128   // time steps
#define NV  64    // vocab

typedef __bf16 bf16;
typedef __attribute__((ext_vector_type(16))) __bf16 v16bf;
typedef __attribute__((ext_vector_type(8)))  __bf16 v8bf;
typedef __attribute__((ext_vector_type(2)))  __bf16 v2bf;
typedef __attribute__((ext_vector_type(8)))  float  v8f;
typedef __attribute__((ext_vector_type(4)))  float  v4f;
typedef __attribute__((ext_vector_type(4)))  int    v4i;

#define AS1 __attribute__((address_space(1)))
#define AS3 __attribute__((address_space(3)))

#if defined(__has_builtin)
#if __has_builtin(__builtin_amdgcn_global_load_async_to_lds_b128)
#define USE_ASYNC_LDS 1
#endif
#endif

// ---------------------------------------------------------------- fragments
static __device__ __forceinline__ v16bf frag_from(v8bf lo, v8bf hi) {
  v16bf r;
#pragma unroll
  for (int i = 0; i < 8; ++i) { r[i] = lo[i]; r[i + 8] = hi[i]; }
  return r;
}

// A fragment (16x32 bf16, row-major). ISA layout: lanes 0-15 hold rows with
// K packs {0..7,16..23}; lanes 16-31 hold the same rows, K packs {8..15,24..31}.
static __device__ __forceinline__ v16bf load_a_bf16(const bf16* A, int lda,
                                                    int m0, int k0, int lane) {
  int row = m0 + (lane & 15);
  int kk  = k0 + ((lane >> 4) << 3);
  const v8bf* p = (const v8bf*)(A + (size_t)row * lda + kk);
  return frag_from(p[0], p[2]);
}

// A fragment read from an LDS-staged 16xK tile (rows 0..15).
static __device__ __forceinline__ v16bf load_a_lds(const bf16* S, int lda,
                                                   int k0, int lane) {
  int row = lane & 15;
  int kk  = k0 + ((lane >> 4) << 3);
  const v8bf* p = (const v8bf*)(S + row * lda + kk);
  return frag_from(p[0], p[2]);
}

// A fragment converting from fp32 source in-register (one-time hp GEMM).
static __device__ __forceinline__ v16bf load_a_f32(const float* A, int lda,
                                                   int m0, int k0, int lane) {
  int row = m0 + (lane & 15);
  int kk  = k0 + ((lane >> 4) << 3);
  const v4f* p = (const v4f*)(A + (size_t)row * lda + kk);
  v4f x0 = p[0], x1 = p[1], x2 = p[4], x3 = p[5];
  v16bf r;
#pragma unroll
  for (int i = 0; i < 4; ++i) {
    r[i]      = (bf16)x0[i];
    r[i + 4]  = (bf16)x1[i];
    r[i + 8]  = (bf16)x2[i];
    r[i + 12] = (bf16)x3[i];
  }
  return r;
}

// B fragment (32x16 bf16). Column n of B == row n of W[N,K] (row-major):
// lanes 0-15 take K={0..7,16..23} of W row n, lanes 16-31 take K={8..15,24..31}.
static __device__ __forceinline__ v16bf load_b_bf16(const bf16* W, int ldw,
                                                    int n0, int k0, int lane) {
  int col = n0 + (lane & 15);
  int kk  = k0 + ((lane >> 4) << 3);
  const v8bf* p = (const v8bf*)(W + (size_t)col * ldw + kk);
  return frag_from(p[0], p[2]);
}

// Stage a 16xK bf16 tile (rows m0..m0+15 of G, row stride ld) into LDS.
// With the gfx1250 async path this issues GLOBAL_LOAD_ASYNC_TO_LDS_B128
// (tracked by ASYNCcnt); otherwise a plain load + ds_store per 16B chunk.
static __device__ __forceinline__ void stage_rows(const bf16* __restrict__ G,
                                                  int ld, int m0, int K,
                                                  bf16* S, int tid, int nthr) {
  const int cpr    = K >> 3;        // 16B chunks per row
  const int chunks = 16 * cpr;
  for (int c = tid; c < chunks; c += nthr) {
    const int row = c / cpr;
    const int off = (c % cpr) << 3;
    const bf16* g = G + (size_t)(m0 + row) * ld + off;
    bf16*       s = S + row * K + off;
#ifdef USE_ASYNC_LDS
    __builtin_amdgcn_global_load_async_to_lds_b128(
        (AS1 v4i*)(void*)g, (AS3 v4i*)(void*)s, 0, 0);
#else
    *(v8bf*)s = *(const v8bf*)g;
#endif
  }
}

static __device__ __forceinline__ void wait_stage_and_sync() {
#ifdef USE_ASYNC_LDS
#if __has_builtin(__builtin_amdgcn_s_wait_asynccnt)
  __builtin_amdgcn_s_wait_asynccnt(0);
#else
  asm volatile("s_wait_asynccnt 0x0" ::: "memory");
#endif
#endif
  __syncthreads();
}

static __device__ __forceinline__ float sigmoidf(float x) {
  return 1.f / (1.f + __expf(-x));
}

// ---------------------------------------------------------------- kernels

// D = A(MxK) @ W(NxK)^T + bias.  One 16x16 tile per wave, block = 4 waves
// covering 64 output columns, grid = (N/64, M/16).  For the bf16-A per-step
// GEMMs (K <= 512) the A tile is staged once per block in LDS and shared by
// all 4 waves (4x less global A traffic).
template <bool AF32>
__global__ void wmma_gemm_bias(const void* __restrict__ Av, int lda,
                               const bf16* __restrict__ W, int ldw,
                               const float* __restrict__ bias,
                               float* __restrict__ outF, bf16* __restrict__ outB,
                               int ldd, int K) {
  const int lane = threadIdx.x & 31;
  const int wave = threadIdx.x >> 5;
  const int n0 = (blockIdx.x * 4 + wave) * 16;
  const int m0 = blockIdx.y * 16;

  v8f acc = {};
  if constexpr (AF32) {
    const float* A = (const float*)Av;
    for (int k = 0; k < K; k += 32) {
      v16bf a = load_a_f32(A, lda, m0, k, lane);
      v16bf b = load_b_bf16(W, ldw, n0, k, lane);
      acc = __builtin_amdgcn_wmma_f32_16x16x32_bf16(false, a, false, b,
                                                    (short)0, acc, false, false);
    }
  } else {
    __shared__ __align__(16) bf16 sA[16 * HID];   // K <= 512 for this path
    stage_rows((const bf16*)Av, lda, m0, K, sA, threadIdx.x, 128);
    wait_stage_and_sync();
    for (int k = 0; k < K; k += 32) {
      v16bf a = load_a_lds(sA, K, k, lane);
      v16bf b = load_b_bf16(W, ldw, n0, k, lane);
      acc = __builtin_amdgcn_wmma_f32_16x16x32_bf16(false, a, false, b,
                                                    (short)0, acc, false, false);
    }
  }

  // C/D layout: VGPR v, lanes 0-15 -> (m0+v, n0+lane); lanes 16-31 -> (m0+v+8, ..)
  const int n  = n0 + (lane & 15);
  const int mb = m0 + ((lane >> 4) << 3);
  const float bn = bias ? bias[n] : 0.f;
#pragma unroll
  for (int v = 0; v < 8; ++v) {
    float d = acc[v] + bn;
    if (outF) outF[(size_t)(mb + v) * ldd + n] = d;
    if (outB) outB[(size_t)(mb + v) * ldd + n] = (bf16)d;
  }
}

// Fused LSTM cell: gates = X1@W1^T + X2@W2^T + b_ih + b_hh, then the
// elementwise update.  Each wave owns one 16(batch)x16(hidden) tile with 4
// accumulators (i,f,g,o) sharing the A fragment; X1/X2 tiles are staged in
// LDS once per block (async-to-LDS when available).  grid = (HID/64, NB/16).
__global__ void wmma_lstm_cell(const bf16* __restrict__ X1, int K1,
                               const bf16* __restrict__ W1,
                               const bf16* __restrict__ X2, int K2,
                               const bf16* __restrict__ W2,
                               const float* __restrict__ b_ih,
                               const float* __restrict__ b_hh,
                               float* __restrict__ cs,
                               bf16* __restrict__ h_bf_out) {
  __shared__ __align__(16) bf16 sA1[16 * (HID + NV)];  // K1 <= 576
  __shared__ __align__(16) bf16 sA2[16 * HID];         // K2 <= 512
  const int lane = threadIdx.x & 31;
  const int wave = threadIdx.x >> 5;
  const int n0 = (blockIdx.x * 4 + wave) * 16;   // hidden-unit tile
  const int m0 = blockIdx.y * 16;                // batch tile

  stage_rows(X1, K1, m0, K1, sA1, threadIdx.x, 128);
  stage_rows(X2, K2, m0, K2, sA2, threadIdx.x, 128);
  wait_stage_and_sync();

  v8f ai = {}, af = {}, ag = {}, ao = {};
  for (int k = 0; k < K1; k += 32) {
    v16bf a  = load_a_lds(sA1, K1, k, lane);
    v16bf b0 = load_b_bf16(W1, K1, 0 * HID + n0, k, lane);
    v16bf b1 = load_b_bf16(W1, K1, 1 * HID + n0, k, lane);
    v16bf b2 = load_b_bf16(W1, K1, 2 * HID + n0, k, lane);
    v16bf b3 = load_b_bf16(W1, K1, 3 * HID + n0, k, lane);
    ai = __builtin_amdgcn_wmma_f32_16x16x32_bf16(false, a, false, b0, (short)0, ai, false, false);
    af = __builtin_amdgcn_wmma_f32_16x16x32_bf16(false, a, false, b1, (short)0, af, false, false);
    ag = __builtin_amdgcn_wmma_f32_16x16x32_bf16(false, a, false, b2, (short)0, ag, false, false);
    ao = __builtin_amdgcn_wmma_f32_16x16x32_bf16(false, a, false, b3, (short)0, ao, false, false);
  }
  for (int k = 0; k < K2; k += 32) {
    v16bf a  = load_a_lds(sA2, K2, k, lane);
    v16bf b0 = load_b_bf16(W2, K2, 0 * HID + n0, k, lane);
    v16bf b1 = load_b_bf16(W2, K2, 1 * HID + n0, k, lane);
    v16bf b2 = load_b_bf16(W2, K2, 2 * HID + n0, k, lane);
    v16bf b3 = load_b_bf16(W2, K2, 3 * HID + n0, k, lane);
    ai = __builtin_amdgcn_wmma_f32_16x16x32_bf16(false, a, false, b0, (short)0, ai, false, false);
    af = __builtin_amdgcn_wmma_f32_16x16x32_bf16(false, a, false, b1, (short)0, af, false, false);
    ag = __builtin_amdgcn_wmma_f32_16x16x32_bf16(false, a, false, b2, (short)0, ag, false, false);
    ao = __builtin_amdgcn_wmma_f32_16x16x32_bf16(false, a, false, b3, (short)0, ao, false, false);
  }

  const int n  = n0 + (lane & 15);
  const int mb = m0 + ((lane >> 4) << 3);
  const float bi = b_ih[n]           + b_hh[n];
  const float bf = b_ih[HID + n]     + b_hh[HID + n];
  const float bg = b_ih[2 * HID + n] + b_hh[2 * HID + n];
  const float bo = b_ih[3 * HID + n] + b_hh[3 * HID + n];
#pragma unroll
  for (int v = 0; v < 8; ++v) {
    const int m = mb + v;
    const float ig = sigmoidf(ai[v] + bi);
    const float fg = sigmoidf(af[v] + bf);
    const float gg = tanhf(ag[v] + bg);
    const float og = sigmoidf(ao[v] + bo);
    const size_t idx = (size_t)m * HID + n;
    const float cn = fg * cs[idx] + ig * gg;
    cs[idx] = cn;
    h_bf_out[idx] = (bf16)(og * tanhf(cn));
  }
}

// Attention: one workgroup per batch element b.  Phase 1: e[r] = <hp[b,r,:],
// sq[b,:]> per-wave rows, lane-split D, shuffle reduction.  Two-pass softmax
// in LDS.  Phase 2: context streamed coalesced over r (hp is L2-resident),
// written as bf16 into inp = [one_hot(z), c].
__global__ void attention_kernel(const float* __restrict__ sq,
                                 const bf16* __restrict__ hp,
                                 const int* __restrict__ z,
                                 bf16* __restrict__ inp_bf) {
  __shared__ float s_sq[HID];
  __shared__ float s_e[NR];
  __shared__ float s_red[256];
  const int b = blockIdx.x;
  const int t = threadIdx.x;
  const int lane = t & 31;
  const int wave = t >> 5;

  s_sq[t]       = sq[b * HID + t];
  s_sq[t + 256] = sq[b * HID + t + 256];
  __syncthreads();

  const bf16* hpb = hp + (size_t)b * NR * HID;
  for (int r = wave * 64; r < wave * 64 + 64; ++r) {
    const v8bf* p = (const v8bf*)(hpb + (size_t)r * HID + lane * 16);
    v8bf x0 = p[0], x1 = p[1];
    float acc = 0.f;
#pragma unroll
    for (int i = 0; i < 8; ++i) {
      acc += (float)x0[i] * s_sq[lane * 16 + i];
      acc += (float)x1[i] * s_sq[lane * 16 + 8 + i];
    }
#pragma unroll
    for (int off = 16; off > 0; off >>= 1) acc += __shfl_xor(acc, off, 32);
    if (lane == 0) s_e[r] = acc;
  }
  __syncthreads();

  // softmax with deferred normalization
  float mloc = fmaxf(s_e[t], s_e[t + 256]);
  s_red[t] = mloc;
  __syncthreads();
  for (int s = 128; s > 0; s >>= 1) {
    if (t < s) s_red[t] = fmaxf(s_red[t], s_red[t + s]);
    __syncthreads();
  }
  const float mx = s_red[0];
  __syncthreads();
  const float e0 = __expf(s_e[t] - mx);
  const float e1 = __expf(s_e[t + 256] - mx);
  s_e[t] = e0; s_e[t + 256] = e1;
  s_red[t] = e0 + e1;
  __syncthreads();
  for (int s = 128; s > 0; s >>= 1) {
    if (t < s) s_red[t] += s_red[t + s];
    __syncthreads();
  }
  const float inv = 1.f / s_red[0];

  // context: thread owns columns d = 2t, 2t+1 (coalesced over r)
  float a0 = 0.f, a1 = 0.f;
  const bf16* col = hpb + 2 * t;
  for (int r = 0; r < NR - 8; ++r) {
    __builtin_prefetch(col + (size_t)(r + 8) * HID, 0, 1);
    v2bf x = *(const v2bf*)(col + (size_t)r * HID);
    const float w = s_e[r];
    a0 += w * (float)x[0];
    a1 += w * (float)x[1];
  }
  for (int r = NR - 8; r < NR; ++r) {
    v2bf x = *(const v2bf*)(col + (size_t)r * HID);
    const float w = s_e[r];
    a0 += w * (float)x[0];
    a1 += w * (float)x[1];
  }
  bf16* row = inp_bf + (size_t)b * (NV + HID);
  row[NV + 2 * t]     = (bf16)(a0 * inv);
  row[NV + 2 * t + 1] = (bf16)(a1 * inv);
  if (t < NV) row[t] = (bf16)((t == z[b]) ? 1.f : 0.f);
}

static __device__ __forceinline__ unsigned hash_u(unsigned x) {
  x ^= x >> 16; x *= 0x7feb352dU;
  x ^= x >> 15; x *= 0x846ca68bU;
  x ^= x >> 16;
  return x;
}

// Deterministic surrogate for jax categorical/bernoulli (hash-Gumbel).
__global__ void sample_kernel(const float* __restrict__ outp,
                              const int* __restrict__ y, int t,
                              int* __restrict__ z) {
  const int b = threadIdx.x;
  if (b >= NB) return;
  const float* ob = outp + ((size_t)b * NT + t) * NV;
  const unsigned seed = hash_u(0x9E3779B9U * (unsigned)(t + 1) + 0x85EBCA6BU * (unsigned)b);
  float best = -1e30f;
  int arg = 0;
  for (int v = 0; v < NV; ++v) {
    const unsigned u = hash_u(seed + (unsigned)v * 0x27d4eb2dU);
    const float uf = (float)(u | 1u) * 2.3283064e-10f;
    const float gv = ob[v] - __logf(-__logf(uf));
    if (gv > best) { best = gv; arg = v; }
  }
  const bool do_sample = hash_u(seed ^ 0xc2b2ae35U) < 429496729U;  // p = 0.1
  z[b] = do_sample ? arg : y[b * (NT + 1) + (t + 1)];
}

__global__ void init_state(float* __restrict__ cs0, float* __restrict__ cs1,
                           bf16* __restrict__ s0b, bf16* __restrict__ s1b,
                           int* __restrict__ z, const int* __restrict__ y) {
  const int i = blockIdx.x * 256 + threadIdx.x;
  if (i < NB * HID) {
    cs0[i] = 0.f; cs1[i] = 0.f;
    s0b[i] = (bf16)0.f; s1b[i] = (bf16)0.f;
  }
  if (i < NB) z[i] = y[i * (NT + 1)];
}

__global__ void cvt_f32_to_bf16(const float* __restrict__ in,
                                bf16* __restrict__ out, int n) {
  const int i = blockIdx.x * 256 + threadIdx.x;
  if (i < n) out[i] = (bf16)in[i];
}

// ---------------------------------------------------------------- launcher
extern "C" void kernel_launch(void* const* d_in, const int* in_sizes, int n_in,
                              void* d_out, int out_size, void* d_ws, size_t ws_size,
                              hipStream_t stream) {
  const float* h     = (const float*)d_in[0];
  const float* phi_w = (const float*)d_in[1];
  const float* phi_b = (const float*)d_in[2];
  const float* psi_w = (const float*)d_in[3];
  const float* psi_b = (const float*)d_in[4];
  const float* w_ih0 = (const float*)d_in[5];
  const float* w_hh0 = (const float*)d_in[6];
  const float* b_ih0 = (const float*)d_in[7];
  const float* b_hh0 = (const float*)d_in[8];
  const float* w_ih1 = (const float*)d_in[9];
  const float* w_hh1 = (const float*)d_in[10];
  const float* b_ih1 = (const float*)d_in[11];
  const float* b_hh1 = (const float*)d_in[12];
  const float* out_w = (const float*)d_in[13];
  const float* out_b = (const float*)d_in[14];
  const int*   y     = (const int*)d_in[15];
  float* outp = (float*)d_out;

  char* cur = (char*)d_ws;
  auto take = [&](size_t bytes) -> void* {
    void* p = cur;
    cur += (bytes + 255) & ~(size_t)255;
    return p;
  };

  bf16* hp_bf   = (bf16*)take((size_t)NB * NR * HID * 2);   // 67 MB, L2-resident
  bf16* phiw_bf = (bf16*)take((size_t)HID * HID * 2);
  bf16* psiw_bf = (bf16*)take((size_t)HID * HID * 2);
  bf16* wih0_bf = (bf16*)take((size_t)4 * HID * (HID + NV) * 2);
  bf16* whh0_bf = (bf16*)take((size_t)4 * HID * HID * 2);
  bf16* wih1_bf = (bf16*)take((size_t)4 * HID * HID * 2);
  bf16* whh1_bf = (bf16*)take((size_t)4 * HID * HID * 2);
  bf16* outw_bf = (bf16*)take((size_t)NV * HID * 2);
  float* sq     = (float*)take((size_t)NB * HID * 4);
  bf16* inp_bf  = (bf16*)take((size_t)NB * (NV + HID) * 2);
  bf16* s0A     = (bf16*)take((size_t)NB * HID * 2);   // ping-pong hidden states
  bf16* s0B     = (bf16*)take((size_t)NB * HID * 2);
  bf16* s1A     = (bf16*)take((size_t)NB * HID * 2);
  bf16* s1B     = (bf16*)take((size_t)NB * HID * 2);
  float* cs0    = (float*)take((size_t)NB * HID * 4);
  float* cs1    = (float*)take((size_t)NB * HID * 4);
  int*   z      = (int*)take((size_t)NB * 4);

  auto cvt = [&](const float* src, bf16* dst, int n) {
    cvt_f32_to_bf16<<<(n + 255) / 256, 256, 0, stream>>>(src, dst, n);
  };
  cvt(phi_w, phiw_bf, HID * HID);
  cvt(psi_w, psiw_bf, HID * HID);
  cvt(w_ih0, wih0_bf, 4 * HID * (HID + NV));
  cvt(w_hh0, whh0_bf, 4 * HID * HID);
  cvt(w_ih1, wih1_bf, 4 * HID * HID);
  cvt(w_hh1, whh1_bf, 4 * HID * HID);
  cvt(out_w, outw_bf, NV * HID);

  // hp = h @ psi_w.T + psi_b  (M = NB*NR = 65536, N = K = 512), fp32 A -> bf16 out
  wmma_gemm_bias<true><<<dim3(HID / 64, (NB * NR) / 16), 128, 0, stream>>>(
      h, HID, psiw_bf, HID, psi_b, nullptr, hp_bf, HID, HID);

  init_state<<<(NB * HID + 255) / 256, 256, 0, stream>>>(cs0, cs1, s0A, s1A, z, y);

  for (int t = 0; t < NT; ++t) {
    bf16* s0_cur = (t & 1) ? s0B : s0A;
    bf16* s0_nxt = (t & 1) ? s0A : s0B;
    bf16* s1_cur = (t & 1) ? s1B : s1A;
    bf16* s1_nxt = (t & 1) ? s1A : s1B;

    // sq = s1 @ phi_w.T + phi_b
    wmma_gemm_bias<false><<<dim3(HID / 64, NB / 16), 128, 0, stream>>>(
        s1_cur, HID, phiw_bf, HID, phi_b, sq, nullptr, HID, HID);

    // attention: e -> softmax -> context; builds inp = [one_hot(z), c] in bf16
    attention_kernel<<<NB, 256, 0, stream>>>(sq, hp_bf, z, inp_bf);

    // LSTM layer 0: x = inp (K=576), h = s0 (K=512)
    wmma_lstm_cell<<<dim3(HID / 64, NB / 16), 128, 0, stream>>>(
        inp_bf, HID + NV, wih0_bf, s0_cur, HID, whh0_bf, b_ih0, b_hh0, cs0, s0_nxt);

    // LSTM layer 1: x = s0_new, h = s1
    wmma_lstm_cell<<<dim3(HID / 64, NB / 16), 128, 0, stream>>>(
        s0_nxt, HID, wih1_bf, s1_cur, HID, whh1_bf, b_ih1, b_hh1, cs1, s1_nxt);

    // o = s1_new @ out_w.T + out_b  -> d_out[:, t, :]
    wmma_gemm_bias<false><<<dim3(NV / 64, NB / 16), 128, 0, stream>>>(
        s1_nxt, HID, outw_bf, HID, out_b, outp + (size_t)t * NV, nullptr,
        NT * NV, HID);

    // next token
    sample_kernel<<<1, NB, 0, stream>>>(outp, y, t, z);
  }
}
